// HeClusTopicModel_5626407158268
// MI455X (gfx1250) — compile-verified, matrix operand order checked by reading, not run on hardware
//
#include <hip/hip_runtime.h>
#include <hip/hip_bf16.h>
#include <math.h>
#include <stdint.h>

typedef __attribute__((ext_vector_type(2)))  float  v2f;
typedef __attribute__((ext_vector_type(8)))  float  v8f;
typedef __attribute__((ext_vector_type(16))) __bf16 v16bf;

#define NTOK   32768   // B*S = 128*256
#define BERT   768
#define HID    512
#define LAT    128
#define VV     4096
#define KC     100
#define VOCABN 30522
#define BB     128     // batch

// async 16B global->LDS copy (gfx1250 GLOBAL_LOAD_ASYNC_TO_LDS_B128, ASYNCcnt)
__device__ __forceinline__ void async_copy_b128(void* lds_dst, const void* gsrc)
{
    uint32_t lds_addr = (uint32_t)(uintptr_t)lds_dst;   // low 32 bits = LDS offset
    asm volatile("global_load_async_to_lds_b128 %0, %1, off"
                 :: "v"(lds_addr), "v"(gsrc) : "memory");
}
// async loads complete in order: waiting <=2 releases the oldest tile-pair
__device__ __forceinline__ void async_wait_le(int n)
{
    if (n == 0) asm volatile("s_wait_asynccnt 0x0" ::: "memory");
    else        asm volatile("s_wait_asynccnt 0x2" ::: "memory");
}

// ---------------------------------------------------------------------------
// Generic fp32 WMMA GEMM:  C[M,N] = act(A[M,K] @ W[K,N] + bias)
// block = 256 threads (8 waves), block tile 64x64, wave tile 32x16,
// double-buffered async global->LDS staging.
// ---------------------------------------------------------------------------
__global__ __launch_bounds__(256)
void gemm_f32_wmma(const float* __restrict__ A, const float* __restrict__ W,
                   const float* __restrict__ bias, float* __restrict__ C,
                   int M, int N, int K, int relu)
{
    __shared__ float As[2][64][20];   // 64 rows x 16 k (80B row stride, 16B aligned)
    __shared__ float Bs[2][16][68];   // 16 k x 64 n    (272B row stride, 16B aligned)

    const int tid  = threadIdx.x;
    const int lane = tid & 31;
    const int wave = tid >> 5;
    const int m0   = blockIdx.y * 64;
    const int n0   = blockIdx.x * 64;
    const int rowBase = (wave >> 2) * 32;   // 0 or 32
    const int colOff  = (wave & 3) * 16;    // 0,16,32,48
    const int lrow = lane & 15;             // row/col within 16
    const int hi   = lane >> 4;             // lane half select

    // staging assignments: one b128 per thread per tile
    const int ar = tid >> 2, ac = (tid & 3) * 4;   // A tile 64x16
    const int br = tid >> 4, bc = (tid & 15) * 4;  // B tile 16x64

    v8f acc0 = {};
    v8f acc1 = {};

    const int nk = K >> 4;
    // prologue: stage tile 0 into buffer 0
    async_copy_b128(&As[0][ar][ac], &A[(size_t)(m0 + ar) * K + ac]);
    async_copy_b128(&Bs[0][br][bc], &W[(size_t)br * N + (n0 + bc)]);

    for (int it = 0; it < nk; ++it) {
        const int cur = it & 1;
        if (it + 1 < nk) {  // stage next tile into the other buffer
            int k1 = (it + 1) << 4;
            async_copy_b128(&As[1 - cur][ar][ac], &A[(size_t)(m0 + ar) * K + (k1 + ac)]);
            async_copy_b128(&Bs[1 - cur][br][bc], &W[(size_t)(k1 + br) * N + (n0 + bc)]);
            if (it + 2 < nk)  // speculative prefetch 2 tiles ahead -> global_prefetch_b8
                __builtin_prefetch(&A[(size_t)(m0 + ar) * K + (k1 + 16 + ac)], 0, 1);
            async_wait_le(2);   // oldest pair (current tile) has landed
        } else {
            async_wait_le(0);
        }
        __syncthreads();

        const float (*Ab)[20] = As[cur];
        const float (*Bb)[68] = Bs[cur];

        #pragma unroll
        for (int kk = 0; kk < 16; kk += 4) {
            // f32 16x16x4 fragment layout (ISA 7.12.2):
            //   A: lane holds row M=lane%16; VGPR0=K(2*hi), VGPR1=K(2*hi+1)
            //   B: lane holds col N=lane%16; VGPR0=K(2*hi), VGPR1=K(2*hi+1)
            int ka = kk + 2 * hi;
            v2f a0, a1, b;
            a0.x = Ab[rowBase      + lrow][ka];
            a0.y = Ab[rowBase      + lrow][ka + 1];
            a1.x = Ab[rowBase + 16 + lrow][ka];
            a1.y = Ab[rowBase + 16 + lrow][ka + 1];
            b.x  = Bb[ka    ][colOff + lrow];
            b.y  = Bb[ka + 1][colOff + lrow];
            acc0 = __builtin_amdgcn_wmma_f32_16x16x4_f32(false, a0, false, b,
                                                         (short)0, acc0, false, false);
            acc1 = __builtin_amdgcn_wmma_f32_16x16x4_f32(false, a1, false, b,
                                                         (short)0, acc1, false, false);
        }
        __syncthreads();   // all waves done reading buf[cur] before it is re-staged
    }

    // C/D layout: VGPR r -> M = r + 8*hi, N = lane%16
    const int col = n0 + colOff + lrow;
    const float bv = bias ? bias[col] : 0.0f;
    #pragma unroll
    for (int r = 0; r < 8; ++r) {
        int row = m0 + rowBase + r + 8 * hi;
        float v0 = acc0[r] + bv;
        float v1 = acc1[r] + bv;
        if (relu) { v0 = fmaxf(v0, 0.0f); v1 = fmaxf(v1, 0.0f); }
        C[(size_t)row * N + col]        = v0;
        C[(size_t)(row + 16) * N + col] = v1;
    }
}

// ---------------------------------------------------------------------------
// Co-occurrence GEMM in bf16 (0/1 indicators are exact in bf16):
//   p_ij = (nz^T @ nz)/B ;  co = p_ij / (p_i[m] * p_i[n])
// A = nz^T [VV x 128] row-major, Bm = nz [128 x VV] row-major
// ---------------------------------------------------------------------------
__global__ __launch_bounds__(256)
void cooc_bf16_wmma(const __bf16* __restrict__ Anz, const __bf16* __restrict__ Bnz,
                    const float* __restrict__ p_i, float* __restrict__ CO)
{
    __shared__ __bf16 As[2][64][40];   // 64 rows x 32 k (80B stride)
    __shared__ __bf16 Bs[2][32][72];   // 32 k x 64 n    (144B stride)

    const int tid  = threadIdx.x;
    const int lane = tid & 31;
    const int wave = tid >> 5;
    const int m0   = blockIdx.y * 64;
    const int n0   = blockIdx.x * 64;
    const int rowBase = (wave >> 2) * 32;
    const int colOff  = (wave & 3) * 16;
    const int lrow = lane & 15;
    const int hi   = lane >> 4;

    const int ar = tid >> 2, ac = (tid & 3) * 8;   // A tile 64x32 bf16, 16B chunks
    const int br = tid >> 3, bc = (tid & 7) * 8;   // B tile 32x64 bf16, 16B chunks

    v8f acc0 = {};
    v8f acc1 = {};

    const int nk = BB / 32;   // 4
    async_copy_b128(&As[0][ar][ac], &Anz[(size_t)(m0 + ar) * BB + ac]);
    async_copy_b128(&Bs[0][br][bc], &Bnz[(size_t)br * VV + (n0 + bc)]);

    for (int it = 0; it < nk; ++it) {
        const int cur = it & 1;
        if (it + 1 < nk) {
            int k1 = (it + 1) * 32;
            async_copy_b128(&As[1 - cur][ar][ac], &Anz[(size_t)(m0 + ar) * BB + (k1 + ac)]);
            async_copy_b128(&Bs[1 - cur][br][bc], &Bnz[(size_t)(k1 + br) * VV + (n0 + bc)]);
            async_wait_le(2);
        } else {
            async_wait_le(0);
        }
        __syncthreads();

        const __bf16 (*Ab)[40] = As[cur];
        const __bf16 (*Bb)[72] = Bs[cur];

        // 16-bit A 16x32 layout (ISA 7.12.2): VGPR0..3 = K 0..7 (+8 if hi lane),
        // VGPR4..7 = K 16..23 (+8 if hi lane); B mirrored by column.
        v16bf a0, a1, b;
        #pragma unroll
        for (int i = 0; i < 8; ++i) {
            int kb = ((i < 4) ? (2 * i) : (16 + 2 * (i - 4))) + 8 * hi;
            a0[2 * i]     = Ab[rowBase      + lrow][kb];
            a0[2 * i + 1] = Ab[rowBase      + lrow][kb + 1];
            a1[2 * i]     = Ab[rowBase + 16 + lrow][kb];
            a1[2 * i + 1] = Ab[rowBase + 16 + lrow][kb + 1];
            b[2 * i]      = Bb[kb    ][colOff + lrow];
            b[2 * i + 1]  = Bb[kb + 1][colOff + lrow];
        }
        acc0 = __builtin_amdgcn_wmma_f32_16x16x32_bf16(false, a0, false, b,
                                                       (short)0, acc0, false, false);
        acc1 = __builtin_amdgcn_wmma_f32_16x16x32_bf16(false, a1, false, b,
                                                       (short)0, acc1, false, false);
        __syncthreads();
    }

    #pragma unroll
    for (int r = 0; r < 8; ++r) {
        int row = m0 + rowBase + r + 8 * hi;
        int col = n0 + colOff + lrow;
        float pij0 = acc0[r] * (1.0f / BB);
        float pij1 = acc1[r] * (1.0f / BB);
        CO[(size_t)row * VV + col]        = pij0 / (p_i[row]      * p_i[col]);
        CO[(size_t)(row + 16) * VV + col] = pij1 / (p_i[row + 16] * p_i[col]);
    }
}

// ---------------------------------------------------------------------------
// Small helper kernels
// ---------------------------------------------------------------------------
__global__ void zero_f32(float* p, int n)
{
    int i = blockIdx.x * 256 + threadIdx.x;
    if (i < n) p[i] = 0.0f;
}

// L2-normalize rows of a [rows x 128] matrix (grid = rows, block = 128)
__global__ __launch_bounds__(128)
void normalize_rows128(const float* __restrict__ in, float* __restrict__ out)
{
    __shared__ float red[128];
    int row = blockIdx.x, t = threadIdx.x;
    float v = in[(size_t)row * LAT + t];
    red[t] = v * v;
    __syncthreads();
    #pragma unroll
    for (int s = 64; s > 0; s >>= 1) {
        if (t < s) red[t] += red[t + s];
        __syncthreads();
    }
    out[(size_t)row * LAT + t] = v / sqrtf(red[0]);
}

// segment-sum of z rows by token id (grid = NTOK, block = 128)
__global__ __launch_bounds__(128)
void segsum_kernel(const int* __restrict__ ids, const float* __restrict__ z,
                   float* __restrict__ seg, float* __restrict__ freq)
{
    int i = blockIdx.x, t = threadIdx.x;
    int id = ids[i];
    atomicAdd(&seg[(size_t)id * LAT + t], z[(size_t)i * LAT + t]);
    if (t == 0) atomicAdd(&freq[id], 1.0f);
}

__global__ __launch_bounds__(128)
void avg_div_kernel(float* __restrict__ seg, const float* __restrict__ freq)
{
    int v = blockIdx.x, t = threadIdx.x;
    seg[(size_t)v * LAT + t] /= fmaxf(freq[v], 1.0f);
}

// build bf16 indicator matrices + p_i (grid = VV, block = 128)
__global__ __launch_bounds__(128)
void build_nz_kernel(const float* __restrict__ bow, __bf16* __restrict__ Anz,
                     __bf16* __restrict__ Bnz, float* __restrict__ p_i)
{
    __shared__ float red[128];
    int v = blockIdx.x, b = threadIdx.x;
    float x  = bow[(size_t)b * VOCABN + v];
    float nz = (x != 0.0f) ? 1.0f : 0.0f;
    __bf16 h = (__bf16)nz;
    Anz[(size_t)v * BB + b] = h;   // nz^T  [VV x 128]
    Bnz[(size_t)b * VV + v] = h;   // nz    [128 x VV]
    red[b] = nz;
    __syncthreads();
    #pragma unroll
    for (int s = 64; s > 0; s >>= 1) {
        if (b < s) red[b] += red[b + s];
        __syncthreads();
    }
    if (b == 0) p_i[v] = red[0] * (1.0f / BB);
}

// per-row argmax over centers (grid = VV, block = 128)
__global__ __launch_bounds__(128)
void assign_clusters_kernel(const float* __restrict__ avg, const float* __restrict__ cn,
                            float* __restrict__ ids_f, int* __restrict__ ids_i)
{
    __shared__ float zrow[128];
    __shared__ float bestv[128];
    __shared__ int   besti[128];
    int v = blockIdx.x, t = threadIdx.x;
    zrow[t] = avg[(size_t)v * LAT + t];
    __syncthreads();
    float dot = -INFINITY;
    if (t < KC) {
        float s = 0.0f;
        #pragma unroll 8
        for (int k = 0; k < LAT; ++k) s += zrow[k] * cn[(size_t)t * LAT + k];
        dot = s;
    }
    bestv[t] = dot;
    besti[t] = t;
    __syncthreads();
    #pragma unroll
    for (int s = 64; s > 0; s >>= 1) {
        if (t < s) {
            bool take = (bestv[t + s] > bestv[t]) ||
                        (bestv[t + s] == bestv[t] && besti[t + s] < besti[t]);
            if (take) { bestv[t] = bestv[t + s]; besti[t] = besti[t + s]; }
        }
        __syncthreads();
    }
    if (t == 0) { ids_f[v] = (float)besti[0]; ids_i[v] = besti[0]; }
}

// sequential per-center scan (updates to different centers commute;
// order within a center is preserved). grid = KC, block = 128.
__global__ __launch_bounds__(128)
void update_centers_kernel(const float* __restrict__ centers, const float* __restrict__ counts,
                           const float* __restrict__ avg, const int* __restrict__ cids,
                           const float* __restrict__ freq, const float* __restrict__ vw,
                           float* __restrict__ out)
{
    int c = blockIdx.x, t = threadIdx.x;
    float cv  = centers[(size_t)c * LAT + t];
    float cnt = counts[c];
    for (int x = 0; x < VV; ++x) {
        if (cids[x] == c && freq[x] > 0.0f) {
            cnt += 1.0f;
            float eta = vw[x] / cnt;
            cv = (1.0f - eta) * cv + eta * avg[(size_t)x * LAT + t];
        }
    }
    out[(size_t)c * LAT + t] = cv;
}

// ---------------------------------------------------------------------------
// Host launcher
// ---------------------------------------------------------------------------
extern "C" void kernel_launch(void* const* d_in, const int* in_sizes, int n_in,
                              void* d_out, int out_size, void* d_ws, size_t ws_size,
                              hipStream_t stream)
{
    (void)in_sizes; (void)n_in; (void)out_size; (void)ws_size;

    const int*   input_ids  = (const int*)  d_in[0];
    const float* token_embs = (const float*)d_in[3];
    const float* bow        = (const float*)d_in[4];
    const float* enc_w1     = (const float*)d_in[5];
    const float* enc_b1     = (const float*)d_in[6];
    const float* enc_w2     = (const float*)d_in[7];
    const float* enc_b2     = (const float*)d_in[8];
    const float* dec_w1     = (const float*)d_in[9];
    const float* dec_b1     = (const float*)d_in[10];
    const float* dec_w2     = (const float*)d_in[11];
    const float* dec_b2     = (const float*)d_in[12];
    const float* centers    = (const float*)d_in[13];
    const float* counts     = (const float*)d_in[14];
    const float* vocab_w    = (const float*)d_in[15];

    // output layout (floats): co | rec | avg | cluster_ids | new_centers
    float* out_co      = (float*)d_out;
    float* out_rec     = out_co  + (size_t)VV * VV;
    float* out_avg     = out_rec + (size_t)NTOK * BERT;
    float* out_ids     = out_avg + (size_t)VV * LAT;
    float* out_centers = out_ids + VV;

    // workspace layout
    char*   ws    = (char*)d_ws;
    float*  h_buf = (float*)ws;                                  // 32768x512 (also reused as r1)
    float*  z_buf = (float*)(ws + (size_t)67108864);             // 32768x128
    __bf16* Anz   = (__bf16*)(ws + (size_t)83886080);            // 4096x128
    __bf16* Bnz   = (__bf16*)(ws + (size_t)84934656);            // 128x4096
    float*  freq  = (float*)(ws + (size_t)85983232);             // 4096
    float*  p_i   = (float*)(ws + (size_t)85999616);             // 4096
    float*  cn    = (float*)(ws + (size_t)86016000);             // 100x128
    int*    ids_i = (int*)  (ws + (size_t)86067200);             // 4096

    // 0) zero the atomic accumulation targets (d_out/d_ws are poisoned)
    zero_f32<<<dim3((VV * LAT + 255) / 256), 256, 0, stream>>>(out_avg, VV * LAT);
    zero_f32<<<dim3((VV + 255) / 256), 256, 0, stream>>>(freq, VV);

    // 1) h = relu(x @ enc_w1 + b1)
    gemm_f32_wmma<<<dim3(HID / 64, NTOK / 64), 256, 0, stream>>>(
        token_embs, enc_w1, enc_b1, h_buf, NTOK, HID, BERT, 1);
    // 2) z = h @ enc_w2 + b2 ; then L2-normalize rows
    gemm_f32_wmma<<<dim3(LAT / 64, NTOK / 64), 256, 0, stream>>>(
        h_buf, enc_w2, enc_b2, z_buf, NTOK, LAT, HID, 0);
    normalize_rows128<<<dim3(NTOK), 128, 0, stream>>>(z_buf, z_buf);
    // 3) r1 = relu(z @ dec_w1 + b1)   (reuse h_buf)
    gemm_f32_wmma<<<dim3(HID / 64, NTOK / 64), 256, 0, stream>>>(
        z_buf, dec_w1, dec_b1, h_buf, NTOK, HID, LAT, 1);
    // 4) rec = r1 @ dec_w2 + b2
    gemm_f32_wmma<<<dim3(BERT / 64, NTOK / 64), 256, 0, stream>>>(
        h_buf, dec_w2, dec_b2, out_rec, NTOK, BERT, HID, 0);

    // 5) segment sum + averages
    segsum_kernel<<<dim3(NTOK), 128, 0, stream>>>(input_ids, z_buf, out_avg, freq);
    avg_div_kernel<<<dim3(VV), 128, 0, stream>>>(out_avg, freq);

    // 6) co-occurrence matrix (bf16 WMMA, exact for 0/1 data)
    build_nz_kernel<<<dim3(VV), 128, 0, stream>>>(bow, Anz, Bnz, p_i);
    cooc_bf16_wmma<<<dim3(VV / 64, VV / 64), 256, 0, stream>>>(Anz, Bnz, p_i, out_co);

    // 7) cluster assignment (argmax of logits == argmax of softmax)
    normalize_rows128<<<dim3(KC), 128, 0, stream>>>(centers, cn);
    assign_clusters_kernel<<<dim3(VV), 128, 0, stream>>>(out_avg, cn, out_ids, ids_i);

    // 8) online center update scan
    update_centers_kernel<<<dim3(KC), 128, 0, stream>>>(
        centers, counts, out_avg, ids_i, freq, vocab_w, out_centers);
}
